// _QuantLSTMCell_37598143709981
// MI455X (gfx1250) — matrix operand order repaired and checked by reading, not compiled, not run on hardware
//
#include <hip/hip_runtime.h>

// ---------------- problem constants ----------------
#define T_SEQ 256
#define BATCH 128
#define IN_DIM 768
#define H_DIM  768
#define KDIM   768              // == IN_DIM == H_DIM
#define NG     (4 * H_DIM)      // 3072 fused gate columns (i,f,c,o)

#define QACC_S  (8.0f / 127.0f)
#define QSIG_S  (1.0f / 127.0f)
#define QTANH_S (1.0f / 127.0f)
#define QCELL_S (4.0f / 127.0f)
#define QOUT_S  (1.0f / 127.0f)

typedef _Float16 v16h  __attribute__((ext_vector_type(16)));
typedef _Float16 half8 __attribute__((ext_vector_type(8)));
typedef float    v8f   __attribute__((ext_vector_type(8)));
typedef int      v4i   __attribute__((ext_vector_type(4)));

union F16Frag { v16h v; half8 q[2]; };

#define AS1 __attribute__((address_space(1)))
#define AS3 __attribute__((address_space(3)))

// 16-byte global -> LDS asynchronous copy (gfx1250 GLOBAL_LOAD_ASYNC_TO_LDS_B128,
// tracked by ASYNCcnt). Builtin expects (global v4i* src, lds v4i* dst, imm, imm).
__device__ __forceinline__ void async_cp16(void* ldsDst, const void* gSrc) {
#if __has_builtin(__builtin_amdgcn_global_load_async_to_lds_b128)
  void* gs = const_cast<void*>(gSrc);
  __builtin_amdgcn_global_load_async_to_lds_b128(
      (AS1 v4i*)gs, (AS3 v4i*)ldsDst, 0, 0);
#else
  unsigned l = (unsigned)(uintptr_t)ldsDst;           // low 32 bits = LDS offset
  unsigned long long g = (unsigned long long)(uintptr_t)gSrc;
  asm volatile("global_load_async_to_lds_b128 %0, %1, off"
               :: "v"(l), "v"(g) : "memory");
#endif
}
__device__ __forceinline__ void async_wait0() {
#if __has_builtin(__builtin_amdgcn_s_wait_asynccnt)
  __builtin_amdgcn_s_wait_asynccnt(0);
#else
  asm volatile("s_wait_asynccnt 0x0" ::: "memory");
#endif
}

__device__ __forceinline__ float fqf(float x, float s, float inv) {
  // symmetric int8 fake-quant, round-to-nearest-even like jnp.round
  float q = rintf(x * inv);
  q = fminf(127.0f, fmaxf(-128.0f, q));
  return q * s;
}
__device__ __forceinline__ float sigm(float x) { return 1.0f / (1.0f + __expf(-x)); }
__device__ __forceinline__ float tanhfast(float x) {
  float e = __expf(2.0f * x);
  return (e - 1.0f) / (e + 1.0f);
}

// =====================================================================
// Kernel 0: prep — recurrent weights -> f16 ([gate][n][k], k-contiguous,
// WMMA B-matrix friendly), init h(f16) ping buffer and f32 cell scratch.
// =====================================================================
__global__ __launch_bounds__(256) void qlstm_prep(
    const float* __restrict__ whi, const float* __restrict__ whf,
    const float* __restrict__ whc, const float* __restrict__ who,
    const float* __restrict__ h0,  const float* __restrict__ c0,
    _Float16* __restrict__ wh16, _Float16* __restrict__ h16,
    float* __restrict__ cbuf)
{
  int idx = blockIdx.x * blockDim.x + threadIdx.x;
  const int WSZ = H_DIM * H_DIM;
  if (idx < 4 * WSZ) {
    int g = idx / WSZ, r = idx - g * WSZ;
    const float* w = (g == 0) ? whi : (g == 1) ? whf : (g == 2) ? whc : who;
    wh16[idx] = (_Float16)w[r];
  } else {
    int j = idx - 4 * WSZ;
    if (j < BATCH * H_DIM) {
      h16[j] = (_Float16)h0[j];
    } else {
      j -= BATCH * H_DIM;
      cbuf[j] = c0[j];
    }
  }
}

// =====================================================================
// Kernel 1: sequence-wide input projection GEMM (no recurrent dep).
//   P[m, g*H+n] = sum_k X[m,k] * W_g[n,k] + bias_g[n],  m in [0, T*B)
// f16 WMMA, block tile 128x128, K-tile 32; 8 waves in 4x2 grid, each
// computing a 32x64 patch (2x4 wmma accumulators). P stored f16.
// =====================================================================
#define TM 128
#define TN 128
#define TK 32

__global__ __launch_bounds__(256) void qlstm_xproj(
    const float* __restrict__ X,
    const float* __restrict__ Wii, const float* __restrict__ Wif,
    const float* __restrict__ Wic, const float* __restrict__ Wio,
    const float* __restrict__ Bi,  const float* __restrict__ Bf,
    const float* __restrict__ Bc,  const float* __restrict__ Bo,
    _Float16* __restrict__ P)
{
  __shared__ __attribute__((aligned(32))) _Float16 sA[TM][40]; // 80B rows (16|80)
  __shared__ __attribute__((aligned(32))) _Float16 sB[TN][40];

  const int tid  = threadIdx.x;
  const int m0   = blockIdx.y * TM;
  const int n0g  = blockIdx.x * TN;        // global fused column
  const int gate = n0g / H_DIM;            // uniform per block (128 | 768)
  const int n0   = n0g - gate * H_DIM;     // column within this gate
  const float* W  = (gate == 0) ? Wii : (gate == 1) ? Wif : (gate == 2) ? Wic : Wio;
  const float* Bv = (gate == 0) ? Bi  : (gate == 1) ? Bf  : (gate == 2) ? Bc  : Bo;

  const int wave = tid >> 5, lane = tid & 31;
  const int wm = wave & 3, wn = wave >> 2;     // 4x2 wave grid -> 128x128
  const int laneRow = lane & 15, laneHi = lane >> 4;

  // staging map: each thread stages 16 floats of A and 16 floats of B
  const int srow = tid >> 1, scol = (tid & 1) * 16;

  v8f acc[2][4] = {};

  for (int k0 = 0; k0 < KDIM; k0 += TK) {
    // ---- stage A tile (X, f32 -> f16 on the fly) ----
    const float* ga = X + (size_t)(m0 + srow) * KDIM + k0 + scol;
    const float* gb = W + (size_t)(n0 + srow) * KDIM + k0 + scol;
    #pragma unroll
    for (int j = 0; j < 16; j += 8) {
      float4 u = *(const float4*)(ga + j);
      float4 w = *(const float4*)(ga + j + 4);
      half8 h;
      h[0] = (_Float16)u.x; h[1] = (_Float16)u.y;
      h[2] = (_Float16)u.z; h[3] = (_Float16)u.w;
      h[4] = (_Float16)w.x; h[5] = (_Float16)w.y;
      h[6] = (_Float16)w.z; h[7] = (_Float16)w.w;
      *(half8*)&sA[srow][scol + j] = h;
      float4 bu = *(const float4*)(gb + j);
      float4 bw = *(const float4*)(gb + j + 4);
      half8 hb;
      hb[0] = (_Float16)bu.x; hb[1] = (_Float16)bu.y;
      hb[2] = (_Float16)bu.z; hb[3] = (_Float16)bu.w;
      hb[4] = (_Float16)bw.x; hb[5] = (_Float16)bw.y;
      hb[6] = (_Float16)bw.z; hb[7] = (_Float16)bw.w;
      *(half8*)&sB[srow][scol + j] = hb;
    }
    if (k0 + TK < KDIM) {          // gfx1250 global_prefetch_b8
      __builtin_prefetch(ga + TK, 0, 1);
      __builtin_prefetch(gb + TK, 0, 1);
    }
    __syncthreads();

    // ---- fragments per ISA layouts ----
    F16Frag af[2], bf[4];
    #pragma unroll
    for (int i = 0; i < 2; ++i) {   // A 16x32: lanes 0-15 K{0-7,16-23}, 16-31 K{8-15,24-31}
      const _Float16* pa = &sA[wm * 32 + i * 16 + laneRow][laneHi * 8];
      af[i].q[0] = *(const half8*)pa;
      af[i].q[1] = *(const half8*)(pa + 16);
    }
    #pragma unroll
    for (int j = 0; j < 4; ++j) {   // B 32x16: lanes 0-15 K{0-7,16-23}, 16-31 K{8-15,24-31}
      const _Float16* pb = &sB[wn * 64 + j * 16 + laneRow][laneHi * 8];
      bf[j].q[0] = *(const half8*)pb;
      bf[j].q[1] = *(const half8*)(pb + 16);
    }
    #pragma unroll
    for (int i = 0; i < 2; ++i)
      #pragma unroll
      for (int j = 0; j < 4; ++j)
        acc[i][j] = __builtin_amdgcn_wmma_f32_16x16x32_f16(
            false, af[i].v, false, bf[j].v, (short)0, acc[i][j], false, false);
    __syncthreads();
  }

  // ---- epilogue: add bias, store P as f16 ----
  const int en = laneRow, em = laneHi * 8;   // C/D: N=lane&15, M=v+8*(lane>=16)
  #pragma unroll
  for (int i = 0; i < 2; ++i)
    #pragma unroll
    for (int j = 0; j < 4; ++j) {
      int nn = n0 + wn * 64 + j * 16 + en;
      float bias = Bv[nn];
      #pragma unroll
      for (int v = 0; v < 8; ++v) {
        int gm = m0 + wm * 32 + i * 16 + em + v;
        P[(size_t)gm * NG + gate * H_DIM + nn] = (_Float16)(acc[i][j][v] + bias);
      }
    }
}

// =====================================================================
// Kernel 2: one recurrent step (launched T times, graph-captured).
// Block = one 16-column slice of H across ALL 4 gates, all 128 batch rows.
// 8 waves: wave w -> gate w>>1, row-half w&1 (4 wmma M-tiles each).
// Staging uses ASYNC global->LDS copies (h and Wh are f16 already).
// Per-gate accumulators go through LDS so every thread can fuse
// sigmoid/tanh/fake-quant + cell update locally.
// =====================================================================
#define TK2 64

__global__ __launch_bounds__(256) void qlstm_step(
    const _Float16* __restrict__ P,    // pre-offset: row t of [T][B][4H]
    const _Float16* __restrict__ Wh,   // [4][H][K] f16
    const _Float16* __restrict__ hIn,  // [B][H] f16 (ping)
    _Float16* __restrict__ hOut,       // [B][H] f16 (pong)
    float* __restrict__ cBuf,          // [B][H] f32, in-place per element
    float* __restrict__ hiddenOut,     // pre-offset: d_out row t
    float* __restrict__ hLast, float* __restrict__ cLast,
    int isLast)
{
  __shared__ __attribute__((aligned(32))) _Float16 sH[BATCH][72]; // 144B rows (16|144)
  __shared__ __attribute__((aligned(32))) _Float16 sW[4][16][72];
  __shared__ float gAcc[4][BATCH][16];

  const int tid = threadIdx.x;
  const int n0  = blockIdx.x * 16;        // 48 blocks cover H=768
  const int wave = tid >> 5, lane = tid & 31;
  const int g = wave >> 1, mh = wave & 1; // gate, row-half (64 rows)
  const int laneRow = lane & 15, laneHi = lane >> 4;

  const int hrow = tid >> 1, hchunk = (tid & 1) * 32;       // 32 f16 / thread
  const int wg = tid >> 6, wn = (tid >> 2) & 15, wchunk = (tid & 3) * 16;

  v8f acc[4] = {};

  for (int k0 = 0; k0 < KDIM; k0 += TK2) {
    // ---- async-stage h tile (128 x 64 f16) and the 4 gates' weight rows ----
    const _Float16* gh = hIn + (size_t)hrow * H_DIM + k0 + hchunk;
    #pragma unroll
    for (int j = 0; j < 4; ++j)
      async_cp16(&sH[hrow][hchunk + j * 8], gh + j * 8);
    const _Float16* gw = Wh + ((size_t)wg * H_DIM + n0 + wn) * KDIM + k0 + wchunk;
    async_cp16(&sW[wg][wn][wchunk],     gw);
    async_cp16(&sW[wg][wn][wchunk + 8], gw + 8);
    async_wait0();
    __syncthreads();

    #pragma unroll
    for (int kk = 0; kk < 2; ++kk) {     // two K=32 WMMA slabs per stage
      F16Frag bf;
      const _Float16* pb = &sW[g][laneRow][kk * 32 + laneHi * 8];
      bf.q[0] = *(const half8*)pb;
      bf.q[1] = *(const half8*)(pb + 16);
      #pragma unroll
      for (int mt = 0; mt < 4; ++mt) {
        F16Frag af;
        const _Float16* pa = &sH[mh * 64 + mt * 16 + laneRow][kk * 32 + laneHi * 8];
        af.q[0] = *(const half8*)pa;
        af.q[1] = *(const half8*)(pa + 16);
        acc[mt] = __builtin_amdgcn_wmma_f32_16x16x32_f16(
            false, af.v, false, bf.v, (short)0, acc[mt], false, false);
      }
    }
    __syncthreads();
  }

  // spill per-gate accumulators so threads can gather all 4 gates
  #pragma unroll
  for (int mt = 0; mt < 4; ++mt)
    #pragma unroll
    for (int v = 0; v < 8; ++v)
      gAcc[g][mh * 64 + mt * 16 + laneHi * 8 + v][laneRow] = acc[mt][v];
  __syncthreads();

  // fused gate nonlinearity + cell update: 128x16 elems, 8 per thread
  #pragma unroll
  for (int r = 0; r < 8; ++r) {
    int e = (r << 8) + tid;
    int b = e >> 4, n = e & 15;
    size_t pe = (size_t)b * NG + n0 + n;
    float ai  = (float)P[pe + 0 * H_DIM] + gAcc[0][b][n];
    float af_ = (float)P[pe + 1 * H_DIM] + gAcc[1][b][n];
    float ac  = (float)P[pe + 2 * H_DIM] + gAcc[2][b][n];
    float ao  = (float)P[pe + 3 * H_DIM] + gAcc[3][b][n];

    float ig = fqf(sigm(fqf(ai,  QACC_S, 127.0f / 8.0f)), QSIG_S,  127.0f);
    float fg = fqf(sigm(fqf(af_, QACC_S, 127.0f / 8.0f)), QSIG_S,  127.0f);
    float cg = fqf(tanhfast(fqf(ac, QACC_S, 127.0f / 8.0f)), QTANH_S, 127.0f);
    float og = fqf(sigm(fqf(ao,  QACC_S, 127.0f / 8.0f)), QSIG_S,  127.0f);

    size_t he = (size_t)b * H_DIM + n0 + n;
    float cOld  = cBuf[he];
    float fcell = fqf(fg * cOld, QCELL_S, 127.0f / 4.0f);
    float icell = fqf(ig * cg,   QCELL_S, 127.0f / 4.0f);
    float cNew  = fqf(fcell + icell, QCELL_S, 127.0f / 4.0f);
    float hTan  = fqf(tanhfast(cNew), QTANH_S, 127.0f);
    float hNew  = fqf(og * hTan, QOUT_S, 127.0f);

    cBuf[he] = cNew;
    hOut[he] = (_Float16)hNew;
    hiddenOut[he] = hNew;
    if (isLast) { hLast[he] = hNew; cLast[he] = cNew; }
  }
}

// =====================================================================
extern "C" void kernel_launch(void* const* d_in, const int* in_sizes, int n_in,
                              void* d_out, int out_size, void* d_ws, size_t ws_size,
                              hipStream_t stream) {
  (void)in_sizes; (void)n_in; (void)out_size; (void)ws_size;

  const float* X   = (const float*)d_in[0];
  const float* h0  = (const float*)d_in[1];
  const float* c0  = (const float*)d_in[2];
  const float* Wii = (const float*)d_in[3];
  const float* Wif = (const float*)d_in[4];
  const float* Wic = (const float*)d_in[5];
  const float* Wio = (const float*)d_in[6];
  const float* Whi = (const float*)d_in[7];
  const float* Whf = (const float*)d_in[8];
  const float* Whc = (const float*)d_in[9];
  const float* Who = (const float*)d_in[10];
  const float* Bi  = (const float*)d_in[11];
  const float* Bf  = (const float*)d_in[12];
  const float* Bc  = (const float*)d_in[13];
  const float* Bo  = (const float*)d_in[14];

  // workspace carve-up
  char* ws = (char*)d_ws;
  _Float16* Wh16 = (_Float16*)ws;                       // 4*768*768 f16
  size_t off = (size_t)4 * H_DIM * H_DIM * sizeof(_Float16);
  _Float16* hPing = (_Float16*)(ws + off);              // 2 x [B][H] f16
  off += (size_t)2 * BATCH * H_DIM * sizeof(_Float16);
  float* cBuf = (float*)(ws + off);                     // [B][H] f32
  off += (size_t)BATCH * H_DIM * sizeof(float);
  _Float16* P = (_Float16*)(ws + off);                  // [T*B][4H] f16

  // 0) prep
  {
    int total = 4 * H_DIM * H_DIM + 2 * BATCH * H_DIM;
    qlstm_prep<<<(total + 255) / 256, 256, 0, stream>>>(
        Whi, Whf, Whc, Who, h0, c0, Wh16, hPing, cBuf);
  }
  // 1) sequence-wide input projection
  {
    dim3 grid(NG / TN, (T_SEQ * BATCH) / TM);  // 24 x 256
    qlstm_xproj<<<grid, 256, 0, stream>>>(
        X, Wii, Wif, Wic, Wio, Bi, Bf, Bc, Bo, P);
  }
  // 2) recurrence
  float* outHidden = (float*)d_out;
  float* hLast = outHidden + (size_t)T_SEQ * BATCH * H_DIM;
  float* cLast = hLast + (size_t)BATCH * H_DIM;
  for (int t = 0; t < T_SEQ; ++t) {
    qlstm_step<<<H_DIM / 16, 256, 0, stream>>>(
        P + (size_t)t * BATCH * NG, Wh16,
        hPing + (size_t)(t & 1) * BATCH * H_DIM,
        hPing + (size_t)((t + 1) & 1) * BATCH * H_DIM,
        cBuf,
        outHidden + (size_t)t * BATCH * H_DIM,
        hLast, cLast, (t == T_SEQ - 1) ? 1 : 0);
  }
}
